// Qwen3MoeAttention_25048249270607
// MI455X (gfx1250) — matrix operand order repaired and checked by reading, not compiled, not run on hardware
//
#include <hip/hip_runtime.h>
#include <hip/hip_bf16.h>
#include <math.h>

typedef __attribute__((ext_vector_type(16))) __bf16 v16bf;
typedef __attribute__((ext_vector_type(8)))  float  v8f;
typedef int v4i __attribute__((vector_size(16)));
typedef __attribute__((address_space(1))) v4i gv4i;
typedef __attribute__((address_space(3))) v4i lv4i;

union FragBF { v16bf v; unsigned int u[8]; };

#if defined(__gfx1250__) && __has_builtin(__builtin_amdgcn_global_load_async_to_lds_b128)
#define HAS_ASYNC 1
#else
#define HAS_ASYNC 0
#endif

// copy 16 bytes global -> LDS (async DMA path on gfx1250, tracked by ASYNCcnt)
__device__ __forceinline__ void cp_async16(void* lds, const void* g) {
#if HAS_ASYNC
    __builtin_amdgcn_global_load_async_to_lds_b128((gv4i*)(void*)g, (lv4i*)lds, 0, 0);
#else
    *(uint4*)lds = *(const uint4*)g;
#endif
}

__device__ __forceinline__ void async_wait0() {
#if HAS_ASYNC
#if __has_builtin(__builtin_amdgcn_s_wait_asynccnt)
    __builtin_amdgcn_s_wait_asynccnt(0);
#else
    asm volatile("s_wait_asynccnt 0" ::: "memory");
#endif
#endif
}

__device__ __forceinline__ unsigned short f2bf(float f) {
    unsigned int u = __float_as_uint(f);
    u += 0x7fffu + ((u >> 16) & 1u);          // round to nearest even
    return (unsigned short)(u >> 16);
}

// A-fragment (16x32 bf16): lanes 0-15 hold K 0-15, lanes 16-31 hold K 16-31 (contiguous)
__device__ __forceinline__ v16bf load_frag_a(const unsigned short* row, int hi) {
    const unsigned int* p = (const unsigned int*)row + hi * 8;
    FragBF f;
#pragma unroll
    for (int j = 0; j < 8; ++j) f.u[j] = p[j];
    return f.v;
}

// B-fragment (32x16 bf16): lanes 0-15 hold K 0-15, lanes 16-31 hold K 16-31 (contiguous)
__device__ __forceinline__ v16bf load_frag_b(const unsigned short* row, int hi) {
    const unsigned int* p = (const unsigned int*)row + hi * 8;
    FragBF f;
#pragma unroll
    for (int j = 0; j < 8; ++j) f.u[j] = p[j];
    return f.v;
}

__global__ void cvt_f32_bf16_kernel(const float* __restrict__ in,
                                    unsigned short* __restrict__ out, int n) {
    int i = blockIdx.x * blockDim.x + threadIdx.x;
    if (i < n) out[i] = f2bf(in[i]);
}

// Tiled transpose + convert: out[c][r] = bf16(in[r][c]).  R, C multiples of 32.
__global__ __launch_bounds__(256) void cvtT_f32_bf16_kernel(
    const float* __restrict__ in, unsigned short* __restrict__ out, int R, int C) {
    __shared__ float tile[32][33];
    int t = threadIdx.x;
    int r0 = blockIdx.y * 32, c0 = blockIdx.x * 32;
    int tr = t >> 3;          // 0..31
    int tc4 = t & 7;          // 8 groups of 4 columns
#pragma unroll
    for (int j = 0; j < 4; ++j) {
        int c = tc4 * 4 + j;
        tile[tr][c] = in[(size_t)(r0 + tr) * C + c0 + c];
    }
    __syncthreads();
#pragma unroll
    for (int j = 0; j < 4; ++j) {
        int c = tc4 * 4 + j;
        out[(size_t)(c0 + tr) * R + r0 + c] = f2bf(tile[c][tr]);
    }
}

// ------------- Tiled bf16 GEMM, f32 accum: C[M,N] = A[M,K] * BT[N,K]^T -------------
// A row-major [M][K], BT row-major [N][K] (pre-transposed weights).
#define BM 128
#define BN 128
#define BKT 32
#define LDSK (BKT + 8)        // 40 ushorts = 80 B row stride (16B aligned)

__device__ __forceinline__ void stage_tile(unsigned short (*lds)[LDSK],
                                           const unsigned short* __restrict__ G,
                                           size_t gbase, int ldg, int t) {
    // 128 rows x 32 bf16 = 512 x 16B chunks, 2 per thread
#pragma unroll
    for (int i = 0; i < 2; ++i) {
        int c = i * 256 + t;
        int r = c >> 2, cp = (c & 3) * 8;
        cp_async16(&lds[r][cp], G + gbase + (size_t)r * ldg + cp);
    }
}

__global__ __launch_bounds__(256) void gemm_bf16_kernel(
    const unsigned short* __restrict__ A, const unsigned short* __restrict__ BT,
    float* __restrict__ C, int M, int N, int K) {
    __shared__ unsigned short lA[2][BM][LDSK];
    __shared__ unsigned short lB[2][BN][LDSK];
    int t = threadIdx.x;
    int lane = t & 31, w = t >> 5;
    int hi = lane >> 4, ln = lane & 15;
    int wm = w >> 1, wn = w & 1;                 // 4x2 wave grid -> 32x64 per wave
    int m0 = blockIdx.y * BM, n0 = blockIdx.x * BN;

    v8f acc[2][4];
#pragma unroll
    for (int a = 0; a < 2; ++a)
#pragma unroll
        for (int b = 0; b < 4; ++b)
#pragma unroll
            for (int v = 0; v < 8; ++v) acc[a][b][v] = 0.0f;

    stage_tile(lA[0], A, (size_t)m0 * K, K, t);
    stage_tile(lB[0], BT, (size_t)n0 * K, K, t);

    int T = K / BKT;
    for (int tt = 0; tt < T; ++tt) {
        int cur = tt & 1;
        async_wait0();
        __syncthreads();
        if (tt + 1 < T) {                        // prefetch next tile into other buffer
            int k1 = (tt + 1) * BKT;
            stage_tile(lA[cur ^ 1], A, (size_t)m0 * K + k1, K, t);
            stage_tile(lB[cur ^ 1], BT, (size_t)n0 * K + k1, K, t);
        }
        v16bf af[2], bfr[4];
#pragma unroll
        for (int a = 0; a < 2; ++a) af[a] = load_frag_a(&lA[cur][wm * 32 + a * 16 + ln][0], hi);
#pragma unroll
        for (int b = 0; b < 4; ++b) bfr[b] = load_frag_b(&lB[cur][wn * 64 + b * 16 + ln][0], hi);
#pragma unroll
        for (int a = 0; a < 2; ++a)
#pragma unroll
            for (int b = 0; b < 4; ++b)
                acc[a][b] = __builtin_amdgcn_wmma_f32_16x16x32_bf16(
                    false, af[a], false, bfr[b], (short)0, acc[a][b], false, false);
    }
#pragma unroll
    for (int a = 0; a < 2; ++a)
#pragma unroll
        for (int b = 0; b < 4; ++b)
#pragma unroll
            for (int v = 0; v < 8; ++v) {
                int row = m0 + wm * 32 + a * 16 + v + hi * 8;
                int col = n0 + wn * 64 + b * 16 + ln;
                C[(size_t)row * N + col] = acc[a][b][v];
            }
}

// ---------------- RMSNorm + RoPE, f32 -> bf16, per (seq, head) ----------------
__global__ __launch_bounds__(128) void normrope_kernel(
    const float* __restrict__ X, const float* __restrict__ Wn,
    unsigned short* __restrict__ Y, int nheads) {
    int s = blockIdx.x, h = blockIdx.y, d = threadIdx.x;
    const size_t base = ((size_t)s * nheads + h) * 128;
    float x = X[base + d];
    float ss = x * x;
#pragma unroll
    for (int m = 16; m > 0; m >>= 1) ss += __shfl_xor(ss, m, 32);
    __shared__ float partial[4];
    __shared__ float xn[128];
    if ((d & 31) == 0) partial[d >> 5] = ss;
    __syncthreads();
    float tot = partial[0] + partial[1] + partial[2] + partial[3];
    float r = rsqrtf(tot * (1.0f / 128.0f) + 1e-6f);
    float n = x * r * Wn[d];
    xn[d] = n;
    __syncthreads();
    int dh = d & 63;
    float ang = (float)s * __powf(1.0e6f, -(float)dh * (1.0f / 64.0f));
    float c, si;
    __sincosf(ang, &si, &c);
    float other = (d < 64) ? xn[d + 64] : xn[d - 64];
    float o = (d < 64) ? (n * c - other * si) : (n * c + other * si);
    Y[base + d] = f2bf(o);
}

// ---------------- Flash attention: block = 128 query rows x 1 head ----------------
// Q: [S][32*128] bf16, Kc: [S][4*128] bf16, VT: [4*128][S] bf16 (pre-transposed)
__global__ __launch_bounds__(256) void attn_kernel(
    const unsigned short* __restrict__ Q, const unsigned short* __restrict__ Kc,
    const unsigned short* __restrict__ VT, unsigned short* __restrict__ O, int S) {
    __shared__ unsigned short sK[32][136];       // [key][dim], 272 B rows (16B aligned)
    __shared__ unsigned short sVT[128][40];      // [dim][key], 80 B rows (16B aligned)
    __shared__ unsigned short sP[8][16][40];     // per-wave P transpose scratch
    int t = threadIdx.x;
    int lane = t & 31, w = t >> 5;
    int hi = lane >> 4, ln = lane & 15;
    int h = blockIdx.y, kvh = h >> 3;            // 8 Q heads per KV head
    int qbase = blockIdx.x * 128;
    int qrow0 = qbase + w * 16;

    v16bf qf[4];                                 // 16x128 Q tile as 4 A-fragments
    {
        const unsigned short* qrow = Q + (size_t)(qrow0 + ln) * 4096 + (size_t)h * 128;
#pragma unroll
        for (int kc = 0; kc < 4; ++kc) qf[kc] = load_frag_a(qrow + kc * 32, hi);
    }

    float mI[8], lI[8];
    v8f acc[8];
#pragma unroll
    for (int v = 0; v < 8; ++v) { mI[v] = -1e30f; lI[v] = 0.0f; }
#pragma unroll
    for (int nb = 0; nb < 8; ++nb)
#pragma unroll
        for (int v = 0; v < 8; ++v) acc[nb][v] = 0.0f;

    const float scale = 0.08838834764831845f;    // 1/sqrt(128)
    int nkb = (qbase + 128) >> 5;

    for (int kb = 0; kb < nkb; ++kb) {
        int ks = kb * 32;
        // async-stage K tile (32 x 256B) and VT tile (128 x 64B): 2 chunks each / thread
#pragma unroll
        for (int i = 0; i < 2; ++i) {
            int c = i * 256 + t;
            int key = c >> 4, cp = (c & 15) * 8;
            cp_async16(&sK[key][cp], Kc + (size_t)(ks + key) * 512 + (size_t)kvh * 128 + cp);
        }
#pragma unroll
        for (int i = 0; i < 2; ++i) {
            int c = i * 256 + t;
            int d = c >> 2, cp = (c & 3) * 8;
            cp_async16(&sVT[d][cp], VT + ((size_t)kvh * 128 + d) * S + ks + cp);
        }
        async_wait0();
        __syncthreads();

        v8f s0, s1;
#pragma unroll
        for (int v = 0; v < 8; ++v) { s0[v] = 0.0f; s1[v] = 0.0f; }
#pragma unroll
        for (int kc = 0; kc < 4; ++kc) {
            v16bf b0 = load_frag_b(&sK[ln][kc * 32], hi);
            v16bf b1 = load_frag_b(&sK[16 + ln][kc * 32], hi);
            s0 = __builtin_amdgcn_wmma_f32_16x16x32_bf16(false, qf[kc], false, b0, (short)0, s0, false, false);
            s1 = __builtin_amdgcn_wmma_f32_16x16x32_bf16(false, qf[kc], false, b1, (short)0, s1, false, false);
        }

        float p0[8], p1[8];
#pragma unroll
        for (int v = 0; v < 8; ++v) {            // online softmax per row
            int q_idx = qrow0 + v + hi * 8;
            float a0 = ((ks + ln) <= q_idx) ? s0[v] * scale : -1e30f;
            float a1 = ((ks + 16 + ln) <= q_idx) ? s1[v] * scale : -1e30f;
            float tm = fmaxf(a0, a1);
#pragma unroll
            for (int msk = 1; msk < 16; msk <<= 1) tm = fmaxf(tm, __shfl_xor(tm, msk, 16));
            float mNew = fmaxf(mI[v], tm);
            float alpha = __expf(mI[v] - mNew);
            float e0 = __expf(a0 - mNew);
            float e1 = __expf(a1 - mNew);
            float rs = e0 + e1;
#pragma unroll
            for (int msk = 1; msk < 16; msk <<= 1) rs += __shfl_xor(rs, msk, 16);
            lI[v] = lI[v] * alpha + rs;
            mI[v] = mNew;
            p0[v] = e0; p1[v] = e1;
#pragma unroll
            for (int nb = 0; nb < 8; ++nb) acc[nb][v] *= alpha;
        }
        // transpose P through per-wave LDS scratch (DS pipe in-order per wave)
#pragma unroll
        for (int v = 0; v < 8; ++v) {
            sP[w][v + hi * 8][ln]      = f2bf(p0[v]);
            sP[w][v + hi * 8][16 + ln] = f2bf(p1[v]);
        }
        v16bf pf = load_frag_a(&sP[w][ln][0], hi);
#pragma unroll
        for (int nb = 0; nb < 8; ++nb) {
            v16bf vf = load_frag_b(&sVT[nb * 16 + ln][0], hi);
            acc[nb] = __builtin_amdgcn_wmma_f32_16x16x32_bf16(
                false, pf, false, vf, (short)0, acc[nb], false, false);
        }
        __syncthreads();
    }
#pragma unroll
    for (int nb = 0; nb < 8; ++nb)
#pragma unroll
        for (int v = 0; v < 8; ++v) {
            float o = acc[nb][v] / lI[v];
            int row = qrow0 + v + hi * 8;
            O[(size_t)row * 4096 + (size_t)h * 128 + nb * 16 + ln] = f2bf(o);
        }
}

extern "C" void kernel_launch(void* const* d_in, const int* in_sizes, int n_in,
                              void* d_out, int out_size, void* d_ws, size_t ws_size,
                              hipStream_t stream) {
    (void)in_sizes; (void)n_in; (void)out_size; (void)ws_size;
    const float* hidden = (const float*)d_in[0];
    const float* wq  = (const float*)d_in[1];
    const float* wk  = (const float*)d_in[2];
    const float* wv  = (const float*)d_in[3];
    const float* wo  = (const float*)d_in[4];
    const float* qnw = (const float*)d_in[5];
    const float* knw = (const float*)d_in[6];
    float* out = (float*)d_out;

    const int S = 2048, HID = 2048, QD = 4096, KVD = 512;

    char* ws = (char*)d_ws;
    size_t off = 0;
    auto take = [&](size_t bytes) -> void* {
        void* p = ws + off;
        off = (off + bytes + 255) & ~(size_t)255;
        return p;
    };
    unsigned short* hB   = (unsigned short*)take((size_t)S * HID * 2);
    unsigned short* wqT  = (unsigned short*)take((size_t)HID * QD * 2);   // [QD][HID]
    unsigned short* wkT  = (unsigned short*)take((size_t)HID * KVD * 2);  // [KVD][HID]
    unsigned short* wvT  = (unsigned short*)take((size_t)HID * KVD * 2);  // [KVD][HID]
    unsigned short* woT  = (unsigned short*)take((size_t)QD * HID * 2);   // [HID][QD]
    float* qF = (float*)take((size_t)S * QD * 4);
    float* kF = (float*)take((size_t)S * KVD * 4);
    float* vF = (float*)take((size_t)S * KVD * 4);
    unsigned short* qB = (unsigned short*)take((size_t)S * QD * 2);
    unsigned short* kB = (unsigned short*)take((size_t)S * KVD * 2);
    unsigned short* vT = (unsigned short*)take((size_t)S * KVD * 2);      // [KVD][S]
    unsigned short* aB = (unsigned short*)take((size_t)S * QD * 2);

    // converts (plain + transposed)
    cvt_f32_bf16_kernel<<<(S * HID + 255) / 256, 256, 0, stream>>>(hidden, hB, S * HID);
    cvtT_f32_bf16_kernel<<<dim3(QD / 32,  HID / 32), 256, 0, stream>>>(wq, wqT, HID, QD);
    cvtT_f32_bf16_kernel<<<dim3(KVD / 32, HID / 32), 256, 0, stream>>>(wk, wkT, HID, KVD);
    cvtT_f32_bf16_kernel<<<dim3(KVD / 32, HID / 32), 256, 0, stream>>>(wv, wvT, HID, KVD);
    cvtT_f32_bf16_kernel<<<dim3(HID / 32, QD / 32),  256, 0, stream>>>(wo, woT, QD, HID);

    // projections (BT layout)
    gemm_bf16_kernel<<<dim3(QD / 128,  S / 128), 256, 0, stream>>>(hB, wqT, qF, S, QD,  HID);
    gemm_bf16_kernel<<<dim3(KVD / 128, S / 128), 256, 0, stream>>>(hB, wkT, kF, S, KVD, HID);
    gemm_bf16_kernel<<<dim3(KVD / 128, S / 128), 256, 0, stream>>>(hB, wvT, vF, S, KVD, HID);

    normrope_kernel<<<dim3(S, 32), 128, 0, stream>>>(qF, qnw, qB, 32);
    normrope_kernel<<<dim3(S, 4),  128, 0, stream>>>(kF, knw, kB, 4);
    cvtT_f32_bf16_kernel<<<dim3(KVD / 32, S / 32), 256, 0, stream>>>(vF, vT, S, KVD); // [KVD][S]

    attn_kernel<<<dim3(S / 128, 32), 256, 0, stream>>>(qB, kB, vT, aB, S);

    gemm_bf16_kernel<<<dim3(HID / 128, S / 128), 256, 0, stream>>>(aB, woT, out, S, HID, QD);
}
